// Regressor2_15281493639426
// MI455X (gfx1250) — matrix operand order repaired and checked by reading, not compiled, not run on hardware
//
#include <hip/hip_runtime.h>

#define HGT 448
#define WID 160
#define CIN 128
#define KPAD 136      // padded channel stride (f16) for 128-ch activation tiles
#define XRPAD 264     // padded channel stride (f16) for 256-ch x_r tiles
#define NTHREADS 320  // 10 wave32s -> 10 N-tiles of 16 columns
#define NPIX (HGT * WID)

typedef __attribute__((ext_vector_type(16))) _Float16 v16h;
typedef __attribute__((ext_vector_type(8)))  _Float16 v8h;
typedef __attribute__((ext_vector_type(8)))  float    v8f;
typedef __attribute__((ext_vector_type(4)))  float    v4f;

__device__ __forceinline__ float leaky(float v) { return v >= 0.0f ? v : 0.01f * v; }

// A-fragment (16x32 f16, M x K): lane L holds M = L%16; elems 0..7 -> K = hi*8..,
// elems 8..15 -> K = 16+hi*8.. (ISA 05_wmma 16-bit A layout). Weights are fp32
// row-major [M][K=128]; read two contiguous 32B chunks and convert to f16.
__device__ __forceinline__ v16h wfrag(const float* __restrict__ Wrow, int row, int kc, int hi) {
    const float* p = Wrow + row * CIN + kc * 32 + hi * 8;
    v4f x0 = *(const v4f*)(p);
    v4f x1 = *(const v4f*)(p + 4);
    v4f x2 = *(const v4f*)(p + 16);
    v4f x3 = *(const v4f*)(p + 20);
    v16h a;
#pragma unroll
    for (int i = 0; i < 4; ++i) {
        a[i]      = (_Float16)x0[i];
        a[4 + i]  = (_Float16)x1[i];
        a[8 + i]  = (_Float16)x2[i];
        a[12 + i] = (_Float16)x3[i];
    }
    return a;
}

// B-fragment (32x16 f16, K x N): lane L holds column n = L%16; elems 0..7 -> K = hi*8..,
// elems 8..15 -> K = 16+hi*8.. (same K order as A). Activations live in LDS
// as [w][c] f16 with KPAD stride, so this is two 16B LDS reads.
__device__ __forceinline__ v16h bfrag(const _Float16* act, int w, int kc, int hi) {
    const _Float16* p = act + w * KPAD + kc * 32 + hi * 8;
    v8h lo = *(const v8h*)(p);
    v8h hh = *(const v8h*)(p + 16);
    v16h b;
#pragma unroll
    for (int i = 0; i < 8; ++i) { b[i] = lo[i]; b[8 + i] = hh[i]; }
    return b;
}

// One grouped-1x1 layer for this row: out[M x 160] = leaky(W[M x 128] @ in + b),
// per-wave over one 16-column N-tile. C layout: VGPR i, n = lane%16, m = i + 8*hi,
// so the 8 results pack into one contiguous v8h LDS store.
template <int MTILES, int OPAD>
__device__ __forceinline__ void gemm_layer(const float* __restrict__ Wrow,
                                           const float* __restrict__ brow,
                                           const _Float16* in, _Float16* outLDS,
                                           int wcol, int n, int hi) {
    v16h bf[4];
#pragma unroll
    for (int kc = 0; kc < 4; ++kc) bf[kc] = bfrag(in, wcol, kc, hi);
#pragma unroll 2
    for (int mt = 0; mt < MTILES; ++mt) {
        v8f acc = {};
#pragma unroll
        for (int kc = 0; kc < 4; ++kc) {
            v16h a = wfrag(Wrow, mt * 16 + n, kc, hi);
            acc = __builtin_amdgcn_wmma_f32_16x16x32_f16(false, a, false, bf[kc],
                                                         (short)0, acc, false, false);
        }
        const float* bp = brow + mt * 16 + 8 * hi;
        v8h o8;
#pragma unroll
        for (int i = 0; i < 8; ++i) o8[i] = (_Float16)leaky(acc[i] + bp[i]);
        *(v8h*)(outLDS + wcol * OPAD + mt * 16 + 8 * hi) = o8;
    }
}

extern "C" __global__ __launch_bounds__(NTHREADS)
void regressor2_fused(const float* __restrict__ x_in,
                      const float* __restrict__ Wc1, const float* __restrict__ bc1,
                      const float* __restrict__ Wc2, const float* __restrict__ bc2,
                      const float* __restrict__ Wc3, const float* __restrict__ bc3,
                      const float* __restrict__ Wc4, const float* __restrict__ bc4,
                      const float* __restrict__ Wr1, const float* __restrict__ br1,
                      const float* __restrict__ Wm2, const float* __restrict__ bm2,
                      const float* __restrict__ Wm3, const float* __restrict__ bm3,
                      float* __restrict__ out)
{
    const int h    = blockIdx.x;     // one row per workgroup
    const int tid  = threadIdx.x;
    const int wave = tid >> 5;       // N-tile id (0..9)
    const int lane = tid & 31;
    const int n    = lane & 15;      // column-in-tile / A-row-in-tile
    const int hi   = lane >> 4;      // lane half
    const int wcol = wave * 16 + n;  // image column w

    extern __shared__ _Float16 smem[];
    _Float16* ping = smem;                      // 160 x KPAD
    _Float16* pong = smem + WID * KPAD;         // 160 x KPAD
    _Float16* xrb  = smem + 2 * WID * KPAD;     // 160 x XRPAD (x_r, 256 ch)

    // Stage x row: x_in[b=0][c][h][w] -> LDS [w][c] f16 (coalesced over w).
    for (int e = tid; e < CIN * WID; e += NTHREADS) {
        int c = e / WID;
        int w = e - c * WID;
        ping[w * KPAD + c] = (_Float16)x_in[(c * HGT + h) * WID + w];
    }
    __syncthreads();
    // After this point every wave touches only its own 16 columns -> no barriers.

    // x_r = leaky(W_reg1 @ x + b): 256 outputs -> xrb
    gemm_layer<16, XRPAD>(Wr1 + (size_t)h * 256 * CIN, br1 + h * 256, ping, xrb, wcol, n, hi);
    // classifier chain (128 -> 128 -> 128 -> 128)
    gemm_layer<8, KPAD>(Wc1 + (size_t)h * 128 * CIN, bc1 + h * 128, ping, pong, wcol, n, hi);
    gemm_layer<8, KPAD>(Wc2 + (size_t)h * 128 * CIN, bc2 + h * 128, pong, ping, wcol, n, hi);
    gemm_layer<8, KPAD>(Wc3 + (size_t)h * 128 * CIN, bc3 + h * 128, ping, pong, wcol, n, hi);

    // cl4: 257 logits; fold argmax over classes 0..255 into the M loop, keep mask.
    const float* W4 = Wc4 + (size_t)h * 257 * CIN;
    const float* b4 = bc4 + (size_t)h * 257;
    v16h bf[4];
#pragma unroll
    for (int kc = 0; kc < 4; ++kc) bf[kc] = bfrag(pong, wcol, kc, hi);

    float best  = -3.0e38f;
    int   bidx  = 0;
    float maskv = 0.0f;
#pragma unroll 2
    for (int mt = 0; mt < 17; ++mt) {
        v8f acc = {};
#pragma unroll
        for (int kc = 0; kc < 4; ++kc) {
            int row = mt * 16 + n;
            if (row > 256) row = 256;  // clamp OOB rows in the padded last tile
            v16h a = wfrag(W4, row, kc, hi);
            acc = __builtin_amdgcn_wmma_f32_16x16x32_f16(false, a, false, bf[kc],
                                                         (short)0, acc, false, false);
        }
        if (mt < 16) {
#pragma unroll
            for (int i = 0; i < 8; ++i) {
                int m = mt * 16 + 8 * hi + i;        // ascending -> first-max ties
                float v = acc[i] + b4[m];
                if (v > best) { best = v; bidx = m; }
            }
        } else if (hi == 0) {
            maskv = acc[0] + b4[256];                // m == 256 lives in VGPR0/lo-lanes
        }
    }
    {   // combine the two lane halves (prefer lower index on ties, like jnp.argmax)
        float ob = __shfl_xor(best, 16);
        int   oi = __shfl_xor(bidx, 16);
        if (ob > best || (ob == best && oi < bidx)) { best = ob; bidx = oi; }
    }
    const int ind = bidx;

    // Per-pixel gathered matvec: y = leaky(x_r . W_cm2[sc] + b), r = y . W_cm3[ir] + b.
    // Split K=256 across the lane pair that owns this column's x_r slice in LDS.
    const int scidx = h * 16  + (ind >> 4);
    const int ir    = h * 256 + ind;
    float acc2[8];
#pragma unroll
    for (int o = 0; o < 8; ++o) acc2[o] = 0.0f;
    const _Float16* xp = xrb + wcol * XRPAD + hi * 128;
    const float* w2 = Wm2 + (size_t)scidx * (256 * 8) + (size_t)hi * (128 * 8);
    for (int c0 = 0; c0 < 128; c0 += 8) {
        v8h xv = *(const v8h*)(xp + c0);
#pragma unroll
        for (int j = 0; j < 8; ++j) {
            float xf = (float)xv[j];
            const float* wp = w2 + (c0 + j) * 8;
            v4f wa = *(const v4f*)(wp);
            v4f wb = *(const v4f*)(wp + 4);
#pragma unroll
            for (int o = 0; o < 4; ++o) {
                acc2[o]     += xf * wa[o];
                acc2[4 + o] += xf * wb[o];
            }
        }
    }
    float r = 0.0f;
#pragma unroll
    for (int o = 0; o < 8; ++o) {
        float t = acc2[o] + __shfl_xor(acc2[o], 16);   // both halves get the sum
        t = leaky(t + bm2[scidx * 8 + o]);
        r += t * Wm3[(size_t)ir * 8 + o];
    }
    r += bm3[ir];

    if (hi == 0) {
        out[h * WID + wcol]        = ((float)ind + r) * (1.0f / 256.0f);  // x_real
        out[NPIX + h * WID + wcol] = leaky(maskv);                        // mask
    }
}

extern "C" void kernel_launch(void* const* d_in, const int* in_sizes, int n_in,
                              void* d_out, int out_size, void* d_ws, size_t ws_size,
                              hipStream_t stream) {
    (void)in_sizes; (void)n_in; (void)out_size; (void)d_ws; (void)ws_size;
    const float* x_in = (const float*)d_in[0];
    const float* Wc1  = (const float*)d_in[1];
    const float* bc1  = (const float*)d_in[2];
    const float* Wc2  = (const float*)d_in[3];
    const float* bc2  = (const float*)d_in[4];
    const float* Wc3  = (const float*)d_in[5];
    const float* bc3  = (const float*)d_in[6];
    const float* Wc4  = (const float*)d_in[7];
    const float* bc4  = (const float*)d_in[8];
    const float* Wr1  = (const float*)d_in[9];
    const float* br1  = (const float*)d_in[10];
    const float* Wm2  = (const float*)d_in[11];
    const float* bm2  = (const float*)d_in[12];
    const float* Wm3  = (const float*)d_in[13];
    const float* bm3  = (const float*)d_in[14];
    float* out = (float*)d_out;

    const size_t shmem = (size_t)(2 * WID * KPAD + WID * XRPAD) * sizeof(_Float16); // 171,520 B
    // Allow >64KB dynamic LDS (320KB/WGP on gfx1250); idempotent, capture-safe.
    (void)hipFuncSetAttribute((const void*)regressor2_fused,
                              hipFuncAttributeMaxDynamicSharedMemorySize, (int)shmem);

    regressor2_fused<<<dim3(HGT), dim3(NTHREADS), shmem, stream>>>(
        x_in, Wc1, bc1, Wc2, bc2, Wc3, bc3, Wc4, bc4, Wr1, br1, Wm2, bm2, Wm3, bm3, out);
}
